// QHDSpatialBlock_82497731821687
// MI455X (gfx1250) — compile-verified, matrix operand order checked by reading, not run on hardware
//
#include <hip/hip_runtime.h>

typedef __attribute__((ext_vector_type(16))) _Float16 v16h;
typedef __attribute__((ext_vector_type(8)))  _Float16 v8h;
typedef __attribute__((ext_vector_type(8)))  float    v8f;
typedef __attribute__((ext_vector_type(4)))  float    v4f;
typedef __attribute__((ext_vector_type(2)))  float    v2f;

#define BQ   2
#define LQ   2048
#define INQ  512
#define HDQ  4096
#define NSQ  16
#define PIF  3.14159265358979323846f

// ---------------------------------------------------------------------------
// Kernel 0: dt = softplus(dt_proj); dt_bar; decay ratios r[n]; VQC Born
// weights w0,w1; closed-form Dirichlet band-filter kernel phi[2048].
// ---------------------------------------------------------------------------
__global__ __launch_bounds__(256)
void k_params(const float* __restrict__ dt_proj, const float* __restrict__ a_log,
              const float* __restrict__ amp_r, const float* __restrict__ amp_i,
              const float* __restrict__ rot, const float* __restrict__ wh,
              float* __restrict__ dt, float* __restrict__ rdec,
              float* __restrict__ wv, float* __restrict__ phi)
{
  __shared__ float red[256];
  __shared__ float s_dtbar, s_w0, s_w1;
  int tid = threadIdx.x;
  float acc = 0.f;
  for (int k = tid; k < HDQ; k += 256) {
    float v = dt_proj[k];
    float sp = (v > 20.f) ? v : log1pf(__expf(v));
    dt[k] = sp;
    acc += sp;
  }
  red[tid] = acc; __syncthreads();
  for (int s = 128; s > 0; s >>= 1) { if (tid < s) red[tid] += red[tid + s]; __syncthreads(); }
  if (tid == 0) s_dtbar = red[0] / (float)HDQ;
  __syncthreads();
  float dtbar = s_dtbar;
  if (tid < NSQ) rdec[tid] = __expf(-__expf(a_log[tid]) * dtbar);
  if (tid == 0) {
    const float s = 0.3f * 1.0f;   // ENT_STRENGTH * WALK_TIME
    float W0 = wh[0], W1 = wh[1], W2 = wh[2], W3 = wh[3];
    float A0 = W0*W0 + W1*W2, A1 = W0*W1 + W1*W3;
    float A2 = W2*W0 + W3*W2, A3 = W2*W1 + W3*W3;
    float C0 = A0*W0 + A1*W2, C1 = A0*W1 + A1*W3;
    float C2 = A2*W0 + A3*W2, C3 = A2*W1 + A3*W3;
    float s2 = s*s, s3 = s2*s;
    float Ur0 = 1.f - 0.5f*s2*A0, Ur1 = -0.5f*s2*A1;
    float Ur2 = -0.5f*s2*A2,      Ur3 = 1.f - 0.5f*s2*A3;
    float Ui0 = s*W0 - s3*C0/6.f, Ui1 = s*W1 - s3*C1/6.f;
    float Ui2 = s*W2 - s3*C2/6.f, Ui3 = s*W3 - s3*C3/6.f;
    float ar0 = amp_r[0], ar1 = amp_r[1], ai0 = amp_i[0], ai1 = amp_i[1];
    for (int d = 0; d < 2; ++d) {
      float c0 = __cosf(rot[d*2+0]), sn0 = __sinf(rot[d*2+0]);
      float c1 = __cosf(rot[d*2+1]), sn1 = __sinf(rot[d*2+1]);
      float tr0 = ar0*c0 - ai0*sn0, ti0 = ar0*sn0 + ai0*c0;
      float tr1 = ar1*c1 - ai1*sn1, ti1 = ar1*sn1 + ai1*c1;
      ar0 = Ur0*tr0 - Ui0*ti0 + Ur1*tr1 - Ui1*ti1;
      ai0 = Ur0*ti0 + Ui0*tr0 + Ur1*ti1 + Ui1*tr1;
      ar1 = Ur2*tr0 - Ui2*ti0 + Ur3*tr1 - Ui3*ti1;
      ai1 = Ur2*ti0 + Ui2*tr0 + Ur3*ti1 + Ui3*tr1;
    }
    float p0 = ar0*ar0 + ai0*ai0, p1 = ar1*ar1 + ai1*ai1;
    float inv = 1.f / (p0 + p1);
    s_w0 = p0*inv; s_w1 = p1*inv;
    wv[0] = s_w0; wv[1] = s_w1;
  }
  __syncthreads();
  float w0 = s_w0, w1 = s_w1, dw = w0 - w1;
  for (int u = tid; u < LQ; u += 256) {
    float lp;
    if (u == 0) {
      lp = 1025.f / 2048.f;
    } else {
      float a = sinf(PIF * (float)u * (1025.f / 2048.f));
      float b = sinf(PIF * (float)u / 2048.f);
      lp = a / (2048.f * b);
    }
    float v = dw * lp;
    if (u == 0) v += w1;          // + w1 * delta[u]
    phi[u] = v;
  }
}

// ---------------------------------------------------------------------------
// One-time f32 -> f16 transpose:  dst[c*R + r] = src[r*C + c]
// ---------------------------------------------------------------------------
__global__ __launch_bounds__(256)
void k_cvt_t(const float* __restrict__ src, _Float16* __restrict__ dst,
             int R, int C)
{
  long id = (long)blockIdx.x * 256 + threadIdx.x;
  if (id >= (long)R * C) return;
  int r = (int)(id % R), c = (int)(id / R);
  dst[id] = (_Float16)src[(long)r * C + c];    // dst[c*R + r], coalesced write
}

// ---------------------------------------------------------------------------
// Weight-fold GEMM: C[M,N] = (A or A^T, optionally per-k scaled) @ B
// B supplied as f16 TRANSPOSED (N x K) so every B fragment is two b128 loads.
// A (non-trans) fragments are four b128 float4 loads. f32 WMMA accumulation.
// ---------------------------------------------------------------------------
template <bool TRANSA, bool HAS_SCALE, bool OUT_F16T>
__global__ __launch_bounds__(256)
void k_gemm_pre(const float* __restrict__ A, const float* __restrict__ scale,
                const _Float16* __restrict__ Bt, float* __restrict__ Cf,
                _Float16* __restrict__ Ch, int M, int N, int K)
{
  int wave = threadIdx.x >> 5, lane = threadIdx.x & 31;
  int mt_n = M >> 4, nt_n = N >> 4;
  int tile = blockIdx.x * 8 + wave;
  if (tile >= mt_n * nt_n) return;          // wave-uniform exit
  int mt = tile / nt_n, nt = tile % nt_n;
  int half = lane >> 4, l15 = lane & 15;
  int m = mt*16 + l15;
  int n = nt*16 + l15;
  int kbase = half * 8;
  const float*    Ap = A  + (TRANSA ? (long)m : (long)m * K);
  const _Float16* Bp = Bt + (long)n * K;
  v8f c = {};
  for (int kk = 0; kk < K; kk += 32) {
    int k0 = kk + kbase;
    v16h a, b;
    if (TRANSA) {
      #pragma unroll
      for (int e = 0; e < 16; ++e) {
        int k = k0 + e + ((e >= 8) ? 8 : 0);
        a[e] = (_Float16)Ap[(long)k * M];
      }
    } else {
      v4f x0 = *(const v4f*)(Ap + k0);
      v4f x1 = *(const v4f*)(Ap + k0 + 4);
      v4f x2 = *(const v4f*)(Ap + k0 + 16);
      v4f x3 = *(const v4f*)(Ap + k0 + 20);
      if (HAS_SCALE) {
        v4f s0 = *(const v4f*)(scale + k0);
        v4f s1 = *(const v4f*)(scale + k0 + 4);
        v4f s2 = *(const v4f*)(scale + k0 + 16);
        v4f s3 = *(const v4f*)(scale + k0 + 20);
        x0 *= s0; x1 *= s1; x2 *= s2; x3 *= s3;
      }
      #pragma unroll
      for (int e = 0; e < 4; ++e) {
        a[e]      = (_Float16)x0[e];
        a[e + 4]  = (_Float16)x1[e];
        a[e + 8]  = (_Float16)x2[e];
        a[e + 12] = (_Float16)x3[e];
      }
    }
    v8h blo = *(const v8h*)(Bp + k0);
    v8h bhi = *(const v8h*)(Bp + k0 + 16);
    #pragma unroll
    for (int e = 0; e < 8; ++e) { b[e] = blo[e]; b[e + 8] = bhi[e]; }
    c = __builtin_amdgcn_wmma_f32_16x16x32_f16(false, a, false, b, (short)0, c, false, false);
  }
  #pragma unroll
  for (int v = 0; v < 8; ++v) {
    int mr = mt*16 + v + 8*half;
    int nc = nt*16 + l15;
    if (OUT_F16T) Ch[(long)nc * M + mr] = (_Float16)c[v];   // transposed f16
    else          Cf[(long)mr * N + nc] = c[v];
  }
}

// ---------------------------------------------------------------------------
// Bt0 = x @ W1  (M=4096, K=512, N=16) on the f32 WMMA path (16x16x4) for
// full precision feeding the SSM recurrence. Stored f16 transposed (c, t).
// ---------------------------------------------------------------------------
__global__ __launch_bounds__(256)
void k_xw1(const float* __restrict__ x, const float* __restrict__ W1f,
           _Float16* __restrict__ Bt0t)
{
  int wave = threadIdx.x >> 5, lane = threadIdx.x & 31;
  int tile = blockIdx.x * 8 + wave;         // 256 M-tiles, single N-tile
  if (tile >= 256) return;
  int half = lane >> 4, l15 = lane & 15;
  int m = tile*16 + l15;
  const float* xp = x + (long)m * INQ;
  v8f c = {};
  for (int kk = 0; kk < INQ; kk += 4) {
    int k0 = kk + half*2;
    v2f a = *(const v2f*)(xp + k0);
    v2f b;
    b[0] = W1f[k0*NSQ + l15];  b[1] = W1f[(k0+1)*NSQ + l15];
    c = __builtin_amdgcn_wmma_f32_16x16x4_f32(false, a, false, b, (short)0, c, false, false);
  }
  int bq = tile >> 7;                       // batch index (tiles don't straddle)
  int col = bq*16 + l15;                    // c = b*16 + state
  #pragma unroll
  for (int v = 0; v < 8; ++v) {
    int t = (tile*16 + v + 8*half) & 2047;
    Bt0t[(long)col * LQ + t] = (_Float16)c[v];
  }
}

// ---------------------------------------------------------------------------
// Band filter as circulant GEMM: BtF[t,c] = sum_s phi[(t-s)&2047] * Bt0[s,c]
// A generated on the fly from phi in LDS; B is transposed f16 -> b128 loads.
// M=2048, N=32, K=2048, f16 WMMA.
// ---------------------------------------------------------------------------
__global__ __launch_bounds__(256)
void k_phi_gemm(const float* __restrict__ phi, const _Float16* __restrict__ Bt0t,
                float* __restrict__ BtF)
{
  __shared__ float sphi[LQ];
  for (int u = threadIdx.x; u < LQ; u += 256) sphi[u] = phi[u];
  __syncthreads();
  int wave = threadIdx.x >> 5, lane = threadIdx.x & 31;
  int tile = blockIdx.x * 8 + wave;         // 128 mt x 2 nt
  if (tile >= 256) return;
  int mt = tile >> 1, nt = tile & 1;
  int half = lane >> 4, l15 = lane & 15;
  int t = mt*16 + l15;
  int n = nt*16 + l15;
  int kbase = half * 8;
  const _Float16* Bp = Bt0t + (long)n * LQ;
  v8f c = {};
  for (int kk = 0; kk < LQ; kk += 32) {
    int k0 = kk + kbase;
    v16h a, b;
    #pragma unroll
    for (int e = 0; e < 16; ++e) {
      int s = k0 + e + ((e >= 8) ? 8 : 0);
      a[e] = (_Float16)sphi[(t - s) & 2047];
    }
    v8h blo = *(const v8h*)(Bp + k0);
    v8h bhi = *(const v8h*)(Bp + k0 + 16);
    #pragma unroll
    for (int e = 0; e < 8; ++e) { b[e] = blo[e]; b[e + 8] = bhi[e]; }
    c = __builtin_amdgcn_wmma_f32_16x16x32_f16(false, a, false, b, (short)0, c, false, false);
  }
  #pragma unroll
  for (int v = 0; v < 8; ++v) {
    int tr = mt*16 + v + 8*half;
    BtF[tr*32 + nt*16 + l15] = c[v];
  }
}

// ---------------------------------------------------------------------------
// Decay conv as first-order scan: h[t] = r_n*h[t-1] + Bt[t].
// One block per (b,n) column; local scan of 8 + Kogge-Stone affine scan.
// ---------------------------------------------------------------------------
__global__ __launch_bounds__(256)
void k_scan(const float* __restrict__ BtF, const float* __restrict__ rdec,
            float* __restrict__ H)
{
  __shared__ float aS[256], bS[256];
  int c = blockIdx.x;                       // 0..31 = b*16+n
  int i = threadIdx.x;
  float r = rdec[c & 15];
  float l[8];
  int t0 = i * 8;
  float acc = 0.f;
  #pragma unroll
  for (int j = 0; j < 8; ++j) { acc = BtF[(t0 + j)*32 + c] + r*acc; l[j] = acc; }
  float q = r*r; q = q*q; q = q*q;          // r^8
  aS[i] = q; bS[i] = acc; __syncthreads();
  for (int d = 1; d < 256; d <<= 1) {
    float ap = 1.f, bp = 0.f;
    if (i >= d) { ap = aS[i - d]; bp = bS[i - d]; }
    __syncthreads();
    if (i >= d) { bS[i] += aS[i]*bp; aS[i] *= ap; }
    __syncthreads();
  }
  float carry = (i > 0) ? bS[i - 1] : 0.f;
  float rp = r;
  #pragma unroll
  for (int j = 0; j < 8; ++j) { H[(t0 + j)*32 + c] = l[j] + rp*carry; rp *= r; }
}

// ---------------------------------------------------------------------------
// Main GEMM: out = x @ W2  (M=4096, K=512, N=512); W2 f16 transposed.
// ---------------------------------------------------------------------------
__global__ __launch_bounds__(256)
void k_main(const float* __restrict__ x, const _Float16* __restrict__ W2t,
            float* __restrict__ out)
{
  int wave = threadIdx.x >> 5, lane = threadIdx.x & 31;
  int tile = blockIdx.x * 8 + wave;         // 256 mt x 32 nt
  if (tile >= 8192) return;
  int mt = tile >> 5, nt = tile & 31;
  int half = lane >> 4, l15 = lane & 15;
  int m = mt*16 + l15, n = nt*16 + l15;
  int kbase = half * 8;
  const float*    xp = x   + (long)m * INQ;
  const _Float16* Bp = W2t + (long)n * INQ;
  v8f c = {};
  for (int kk = 0; kk < INQ; kk += 32) {
    int k0 = kk + kbase;
    if (kk + 64 < INQ)
      __builtin_prefetch(xp + kk + 64, 0, 1);   // global_prefetch_b8
    v4f x0 = *(const v4f*)(xp + k0);
    v4f x1 = *(const v4f*)(xp + k0 + 4);
    v4f x2 = *(const v4f*)(xp + k0 + 16);
    v4f x3 = *(const v4f*)(xp + k0 + 20);
    v16h a, b;
    #pragma unroll
    for (int e = 0; e < 4; ++e) {
      a[e]      = (_Float16)x0[e];
      a[e + 4]  = (_Float16)x1[e];
      a[e + 8]  = (_Float16)x2[e];
      a[e + 12] = (_Float16)x3[e];
    }
    v8h blo = *(const v8h*)(Bp + k0);
    v8h bhi = *(const v8h*)(Bp + k0 + 16);
    #pragma unroll
    for (int e = 0; e < 8; ++e) { b[e] = blo[e]; b[e + 8] = bhi[e]; }
    c = __builtin_amdgcn_wmma_f32_16x16x32_f16(false, a, false, b, (short)0, c, false, false);
  }
  #pragma unroll
  for (int v = 0; v < 8; ++v) {
    int mr = mt*16 + v + 8*half;
    out[(long)mr*INQ + nt*16 + l15] = c[v];
  }
}

// ---------------------------------------------------------------------------
// Rank-16 epilogue: out[m,n] += sum_j H[t, b*16+j] * Mmat[j, n]
// ---------------------------------------------------------------------------
__global__ __launch_bounds__(256)
void k_add_h(const float* __restrict__ H, const float* __restrict__ Mmat,
             float* __restrict__ out)
{
  long id = (long)blockIdx.x * 256 + threadIdx.x;   // 4096*512
  if (id >= (long)4096 * 512) return;
  int m = (int)(id >> 9), n = (int)(id & 511);
  int bq = m >> 11, t = m & 2047;
  const float* hr = H + t*32 + bq*16;
  float acc = out[id];
  #pragma unroll
  for (int j = 0; j < 16; ++j) acc += hr[j] * Mmat[j*512 + n];
  out[id] = acc;
}

// ---------------------------------------------------------------------------
extern "C" void kernel_launch(void* const* d_in, const int* in_sizes, int n_in,
                              void* d_out, int out_size, void* d_ws, size_t ws_size,
                              hipStream_t stream) {
  const float* x        = (const float*)d_in[0];   // (2,2048,512)
  const float* in_proj  = (const float*)d_in[1];   // (512,4096)
  const float* out_proj = (const float*)d_in[2];   // (4096,512)
  const float* a_log    = (const float*)d_in[3];   // (16,)
  const float* b_proj   = (const float*)d_in[4];   // (4096,16)
  const float* c_proj   = (const float*)d_in[5];   // (4096,16)
  const float* dt_proj  = (const float*)d_in[6];   // (4096,)
  const float* skip     = (const float*)d_in[7];   // (4096,)
  const float* amp_r    = (const float*)d_in[8];   // (2,)
  const float* amp_i    = (const float*)d_in[9];   // (2,)
  const float* rot      = (const float*)d_in[10];  // (2,2)
  const float* wh       = (const float*)d_in[11];  // (2,2)
  float* out = (float*)d_out;

  float* ws   = (float*)d_ws;
  float* dt   = ws;                             // 4096
  float* rdec = ws + 4096;                      // 16
  float* wv   = ws + 4112;                      // 16 (2 used)
  float* phi  = ws + 4128;                      // 2048
  float* W1   = ws + 6176;                      // 512*16
  float* Mm   = ws + 14368;                     // 16*512
  _Float16* W2t  = (_Float16*)(ws + 22560);     // 512*512 halves (n-major)
  _Float16* Bt0t = (_Float16*)(ws + 153632);    // 32*2048 halves (c-major)
  float* BtF  = ws + 186400;                    // 2048*32
  float* H    = ws + 251936;                    // 2048*32
  _Float16* OPt = (_Float16*)(ws + 317472);     // 512*4096 halves (n-major)
  _Float16* BPt = (_Float16*)(ws + 1366048);    // 16*4096 halves (n-major)

  // scalars / Dirichlet filter kernel / VQC Born weights
  k_params<<<1, 256, 0, stream>>>(dt_proj, a_log, amp_r, amp_i, rot, wh,
                                  dt, rdec, wv, phi);
  // one-time f16 transposes of the shared B operands
  k_cvt_t<<<8192, 256, 0, stream>>>(out_proj, OPt, HDQ, INQ);   // (4096x512)->(512x4096)
  k_cvt_t<<<256, 256, 0, stream>>>(b_proj, BPt, HDQ, NSQ);      // (4096x16)->(16x4096)
  // W1 = (W_in * dt) @ b_proj                       (512x16, f32)
  k_gemm_pre<false, true, false><<<4, 256, 0, stream>>>(in_proj, dt, BPt, W1, nullptr,
                                                        512, NSQ, HDQ);
  // W2 = (W_in * skip) @ W_out  -> f16 transposed   (512x512)
  k_gemm_pre<false, true, true><<<128, 256, 0, stream>>>(in_proj, skip, OPt, nullptr, W2t,
                                                         512, 512, HDQ);
  // Mmat = c_proj^T @ W_out                         (16x512, f32)
  k_gemm_pre<true, false, false><<<4, 256, 0, stream>>>(c_proj, nullptr, OPt, Mm, nullptr,
                                                        NSQ, 512, HDQ);
  // Bt0 = x @ W1 (f32 WMMA), stored f16 transposed
  k_xw1<<<32, 256, 0, stream>>>(x, W1, Bt0t);
  // band filter (circulant GEMM)
  k_phi_gemm<<<32, 256, 0, stream>>>(phi, Bt0t, BtF);
  // decay conv as first-order scan
  k_scan<<<32, 256, 0, stream>>>(BtF, rdec, H);
  // out = x @ W2
  k_main<<<1024, 256, 0, stream>>>(x, W2t, out);
  // out += H @ Mmat
  k_add_h<<<8192, 256, 0, stream>>>(H, Mm, out);
}
